// VQVAE_52828097740999
// MI455X (gfx1250) — matrix operand order repaired and checked by reading, not compiled, not run on hardware
//
#include <hip/hip_runtime.h>
#include <cmath>

// ---------------------------------------------------------------------------
// VQ-VAE forward for MI455X (gfx1250, wave32, WMMA).
// GEMM-shaped work -> v_wmma_f32_16x16x32_f16 (f32 accumulate).
// fp32 in memory (problem is memory-bound); f16 only in-register, produced by
// the packed hardware converter v_cvt_pk_rtz_f16_f32 (1 op / 2 elements).
// Each wave carries MT=4 M-tiles so the scattered im2col B-gather is done
// once per 4 WMMAs.
// ---------------------------------------------------------------------------

typedef __attribute__((ext_vector_type(16))) _Float16 v16h;
typedef __attribute__((ext_vector_type(8)))  float    v8f;
typedef __attribute__((ext_vector_type(2)))  __fp16   h2;   // cvt_pkrtz result type

union Frag { v16h v; h2 p[8]; };

#define DEV static __device__ __forceinline__

DEV h2 pk(float a, float b) { return __builtin_amdgcn_cvt_pkrtz(a, b); }

DEV v8f wmma_f16(v16h a, v16h b, v8f c) {
  // D = A(16x32) * B(32x16) + C, f32 accumulate
  return __builtin_amdgcn_wmma_f32_16x16x32_f16(false, a, false, b,
                                                (short)0, c, false, false);
}

DEV float lrelu(float x)    { return x > 0.f ? x : 0.01f * x; }
DEV float sigmoidf(float x) { return 1.f / (1.f + __expf(-x)); }

// ---------------------------------------------------------------------------
// conv1: x(64,1,4096) * w(128,1,16) s2 p7 -> lrelu -> h1(64,128,2048). VALU.
// ---------------------------------------------------------------------------
__global__ void conv1_lrelu_k(const float* __restrict__ x,
                              const float* __restrict__ w,
                              float* __restrict__ y) {
  int tid = blockIdx.x * 256 + threadIdx.x;    // 64*128*2048
  int b  = tid >> 18;
  int r  = tid & 262143;
  int co = r >> 11;
  int l  = r & 2047;
  const float* xb = x + (size_t)b * 4096;
  const float* wc = w + co * 16;
  int p0 = 2 * l - 7;
  float s = 0.f;
#pragma unroll
  for (int t = 0; t < 16; ++t) {
    int p = p0 + t;
    if (p >= 0 && p < 4096) s += wc[t] * xb[p];
  }
  y[tid] = lrelu(s);
}

// ---------------------------------------------------------------------------
// im2col-GEMM conv via WMMA with M-tile blocking. A wave owns MT co-tiles
// over one 16-wide n-range (n = b*Lout + l): B gathered once per k-step,
// MT WMMAs issued against MT weight fragments (float2 b64 loads, rows are
// K-contiguous). ACT: 0 none, 1 lrelu. Lout is a power of two.
// ---------------------------------------------------------------------------
template <int KW, int STRIDE, int PAD, int ACT, int MT>
__global__ void conv_gemm_k(const float* __restrict__ X,
                            const float* __restrict__ W,
                            const float* __restrict__ bias,
                            float* __restrict__ Y,
                            int Cin, int Cout, int Lin, int Lout,
                            int logLout, int Ntot) {
  int wid  = blockIdx.x * (blockDim.x >> 5) + (threadIdx.x >> 5);
  int lane = threadIdx.x & 31;
  int tilesMW = Cout >> (4 + (MT == 4 ? 2 : (MT == 2 ? 1 : 0)));
  int tmw = wid % tilesMW, tn = wid / tilesMW;
  int n0 = tn << 4;
  if (n0 >= Ntot) return;                        // wave-uniform
  int mrow = lane & 15, half8 = (lane >> 4) << 3;
  int Ktot = Cin * KW;                           // always a multiple of 32
  int lmask = Lout - 1;
  v8f acc[MT];
#pragma unroll
  for (int mt = 0; mt < MT; ++mt) acc[mt] = {};
  for (int kk = 0; kk < Ktot; kk += 32) {
    float xb[16];
#pragma unroll
    for (int e = 0; e < 16; ++e) {               // gather first (load clause)
      int kg = kk + ((e & 8) << 1) + half8 + (e & 7);  // B: lane = N, element = K
      int ci = kg / KW, t = kg % KW;
      int n  = n0 + mrow;
      int bi = n >> logLout;
      int l  = n & lmask;
      int pos = l * STRIDE - PAD + t;
      xb[e] = (pos >= 0 && pos < Lin) ? X[((size_t)bi * Cin + ci) * Lin + pos]
                                      : 0.f;
    }
    Frag b;
#pragma unroll
    for (int i = 0; i < 8; ++i) b.p[i] = pk(xb[2 * i], xb[2 * i + 1]);
#pragma unroll
    for (int mt = 0; mt < MT; ++mt) {
      int co_a = ((tmw * MT + mt) << 4) + mrow;
      const float* Wr = W + (size_t)co_a * Ktot;
      Frag a;
#pragma unroll
      for (int i = 0; i < 8; ++i) {              // A pairs: consecutive k
        int e = i << 1;
        int k = kk + ((e & 8) << 1) + half8 + (e & 7);
        float2 w2 = *(const float2*)(Wr + k);
        a.p[i] = pk(w2.x, w2.y);
      }
      acc[mt] = wmma_f16(a.v, b.v, acc[mt]);
    }
  }
#pragma unroll
  for (int mt = 0; mt < MT; ++mt)
#pragma unroll
    for (int v = 0; v < 8; ++v) {                // D: reg = M, lane = N
      int co = ((tmw * MT + mt) << 4) + v + half8;
      int n  = n0 + mrow;
      int bi = n >> logLout, l = n & lmask;
      float val = acc[mt][v] + (bias ? bias[co] : 0.f);
      if (ACT == 1) val = lrelu(val);
      Y[((size_t)bi * Cout + co) * Lout + l] = val;
    }
}

// ---------------------------------------------------------------------------
// Transposed conv as conv over the zero-dilated input (JAX lhs_dilation form):
//   Wflip[o][i][t] = Wt[i][o][KW-1-t],  pad_eff = KW-1-PAD.
// Flipped weight pairs are reverse-contiguous -> float2 load + swap.
// Same MT blocking as conv_gemm_k.
// ---------------------------------------------------------------------------
template <int KW, int STRIDE, int PAD, int ACT, int MT>
__global__ void convt_gemm_k(const float* __restrict__ X,
                             const float* __restrict__ Wt,   // (Cin,Cout,KW)
                             float* __restrict__ Y,
                             int Cin, int Cout, int Lin, int Lout,
                             int logLout, int Ntot) {
  const int PADE = KW - 1 - PAD;
  int wid  = blockIdx.x * (blockDim.x >> 5) + (threadIdx.x >> 5);
  int lane = threadIdx.x & 31;
  int tilesMW = Cout >> (4 + (MT == 4 ? 2 : (MT == 2 ? 1 : 0)));
  int tmw = wid % tilesMW, tn = wid / tilesMW;
  int n0 = tn << 4;
  if (n0 >= Ntot) return;
  int mrow = lane & 15, half8 = (lane >> 4) << 3;
  int Ktot = Cin * KW;
  int Ld   = (Lin - 1) * STRIDE + 1;
  int lmask = Lout - 1;
  v8f acc[MT];
#pragma unroll
  for (int mt = 0; mt < MT; ++mt) acc[mt] = {};
  for (int kk = 0; kk < Ktot; kk += 32) {
    float xb[16];
#pragma unroll
    for (int e = 0; e < 16; ++e) {
      int kg = kk + ((e & 8) << 1) + half8 + (e & 7);
      int ciB = kg / KW, tB = kg % KW;
      int n  = n0 + mrow;
      int bi = n >> logLout;
      int l  = n & lmask;
      int pd = l - PADE + tB;
      float v = 0.f;
      if (pd >= 0 && pd < Ld && (pd % STRIDE) == 0)
        v = X[((size_t)bi * Cin + ciB) * Lin + pd / STRIDE];
      xb[e] = v;
    }
    Frag b;
#pragma unroll
    for (int i = 0; i < 8; ++i) b.p[i] = pk(xb[2 * i], xb[2 * i + 1]);
#pragma unroll
    for (int mt = 0; mt < MT; ++mt) {
      int co_a = ((tmw * MT + mt) << 4) + mrow;
      Frag a;
#pragma unroll
      for (int i = 0; i < 8; ++i) {
        int e  = i << 1;
        int k  = kk + ((e & 8) << 1) + half8 + (e & 7);  // even; t even, t+1<KW
        int ci = k / KW, t = k % KW;
        const float* base = Wt + ((size_t)ci * Cout + co_a) * KW;
        float2 w2 = *(const float2*)(base + (KW - 2 - t));
        a.p[i] = pk(w2.y, w2.x);  // elem e = W[KW-1-t], elem e+1 = W[KW-2-t]
      }
      acc[mt] = wmma_f16(a.v, b.v, acc[mt]);
    }
  }
#pragma unroll
  for (int mt = 0; mt < MT; ++mt)
#pragma unroll
    for (int v = 0; v < 8; ++v) {
      int co = ((tmw * MT + mt) << 4) + v + half8;
      int n  = n0 + mrow;
      int bi = n >> logLout, l = n & lmask;
      float val = acc[mt][v];
      if (ACT == 1) val = lrelu(val);
      Y[((size_t)bi * Cout + co) * Lout + l] = val;
    }
}

// ---------------------------------------------------------------------------
// ||emb_row||^2 precompute (1024 rows of 128)
// ---------------------------------------------------------------------------
__global__ void emb_norm_k(const float* __restrict__ emb,
                           float* __restrict__ enorm) {
  int t = blockIdx.x * 256 + threadIdx.x;
  if (t >= 1024) return;
  float s = 0.f;
#pragma unroll 8
  for (int i = 0; i < 128; ++i) { float v = emb[(size_t)t * 128 + i]; s += v * v; }
  enorm[t] = s;
}

// ---------------------------------------------------------------------------
// VQ assign: per wave, 16 latent vectors vs all 1024 embeddings via WMMA.
// d2 = ||e||^2 - 2*z.e ; argmin reduced with a 16-lane shfl_xor tree directly
// on the D-tile layout.  z layout (B=64,D=128,L=512).
// ---------------------------------------------------------------------------
__global__ void vq_assign_k(const float* __restrict__ z,
                            const float* __restrict__ emb,
                            const float* __restrict__ enorm,
                            int* __restrict__ idxOut,
                            float* __restrict__ hist) {
  int wid  = blockIdx.x * 8 + (threadIdx.x >> 5);   // 2048 waves total
  int lane = threadIdx.x & 31;
  int mrow = lane & 15, half8 = (lane >> 4) << 3;
  int vm0 = wid << 4;
  int v   = vm0 + mrow;                             // A row = latent vector
  int b = v >> 9, l = v & 511;
  const float* zb = z + ((size_t)b * 128) * 512 + l;
  Frag a[4];
#pragma unroll
  for (int kb = 0; kb < 4; ++kb) {
    float za[16];
#pragma unroll
    for (int e = 0; e < 16; ++e) {
      int k = (kb << 5) + ((e & 8) << 1) + half8 + (e & 7);
      za[e] = zb[(size_t)k * 512];
    }
#pragma unroll
    for (int i = 0; i < 8; ++i) a[kb].p[i] = pk(za[2 * i], za[2 * i + 1]);
  }
  float best[8]; int bidx[8];
#pragma unroll
  for (int i = 0; i < 8; ++i) { best[i] = 3.4e38f; bidx[i] = 0; }
  for (int e0 = 0; e0 < 1024; e0 += 16) {
    v8f acc = {};
#pragma unroll
    for (int kb = 0; kb < 4; ++kb) {
      int k = (kb << 5) + half8;                    // B: lane = N
      float eb[16];
#pragma unroll
      for (int e = 0; e < 16; ++e)
        eb[e] = emb[(size_t)(e0 + mrow) * 128 + k + ((e & 8) << 1) + (e & 7)];
      Frag bf;
#pragma unroll
      for (int i = 0; i < 8; ++i) bf.p[i] = pk(eb[2 * i], eb[2 * i + 1]);
      acc = wmma_f16(a[kb].v, bf.v, acc);
    }
    int ei = e0 + mrow;                             // this lane's column N
    float en = enorm[ei];
#pragma unroll
    for (int r = 0; r < 8; ++r) {
      float d2 = en - 2.f * acc[r];
      if (d2 < best[r] || (d2 == best[r] && ei < bidx[r])) {
        best[r] = d2; bidx[r] = ei;
      }
    }
  }
#pragma unroll
  for (int off = 1; off < 16; off <<= 1) {          // min over N (16 lanes/half)
#pragma unroll
    for (int r = 0; r < 8; ++r) {
      float ob = __shfl_xor(best[r], off, 32);
      int   oi = __shfl_xor(bidx[r], off, 32);
      if (ob < best[r] || (ob == best[r] && oi < bidx[r])) {
        best[r] = ob; bidx[r] = oi;
      }
    }
  }
  if (mrow == 0) {
#pragma unroll
    for (int r = 0; r < 8; ++r) {
      idxOut[vm0 + r + half8] = bidx[r];
      atomicAdd(&hist[bidx[r]], 1.f);
    }
  }
}

// ---------------------------------------------------------------------------
// Gather quantized vectors into encoded (B,D,L) output + accumulate MSE.
// ---------------------------------------------------------------------------
__global__ void vq_gather_loss_k(const float* __restrict__ z,
                                 const float* __restrict__ emb,
                                 const int* __restrict__ idx,
                                 float* __restrict__ encOut,
                                 float* __restrict__ lossAcc) {
  int tid = blockIdx.x * 256 + threadIdx.x;   // 64*128*512
  int b = tid >> 16;
  int r = tid & 65535;
  int d = r >> 9, l = r & 511;
  int vi = (b << 9) + l;
  float q  = emb[(size_t)idx[vi] * 128 + d];
  float zv = z[tid];
  encOut[tid] = q;
  float df = q - zv;
  __shared__ float red[256];
  red[threadIdx.x] = df * df;
  __syncthreads();
  for (int s = 128; s > 0; s >>= 1) {
    if (threadIdx.x < s) red[threadIdx.x] += red[threadIdx.x + s];
    __syncthreads();
  }
  if (threadIdx.x == 0) atomicAdd(lossAcc, red[0]);
}

// ---------------------------------------------------------------------------
// perplexity + loss scalars
// ---------------------------------------------------------------------------
__global__ void vq_finalize_k(const float* __restrict__ hist,
                              const float* __restrict__ lossAcc,
                              float* __restrict__ perpOut,
                              float* __restrict__ lossOut) {
  __shared__ float red[1024];
  int t = threadIdx.x;
  float avg = hist[t] * (1.f / 32768.f);
  red[t] = -avg * __logf(avg + 1e-10f);
  __syncthreads();
  for (int s = 512; s > 0; s >>= 1) {
    if (t < s) red[t] += red[t + s];
    __syncthreads();
  }
  if (t == 0) {
    *perpOut = __expf(red[0]);
    *lossOut = 1.25f * lossAcc[0] * (1.f / 4194304.f);  // (1+COMMIT)*MSE
  }
}

// ---------------------------------------------------------------------------
// Regression layer 1: hidden(64,512) = flat(64,65536) @ W1^T, WMMA split-K.
// 32 M-tiles x 64 K-segments -> 2048 waves; each wave carries 4 N-tile accums
// so W1 (134 MB) is streamed exactly once (+ prefetch). atomicAdd merge.
// ---------------------------------------------------------------------------
__global__ void reg1_gemm_k(const float* __restrict__ W1,
                            const float* __restrict__ flat,
                            float* __restrict__ hidden) {
  int wid  = blockIdx.x * 8 + (threadIdx.x >> 5);
  int lane = threadIdx.x & 31;
  int mrow = lane & 15, half8 = (lane >> 4) << 3;
  int mt = wid & 31, kseg = wid >> 5;
  int m_a = (mt << 4) + mrow;
  const float* Wr = W1 + (size_t)m_a * 65536;
  v8f acc[4] = {{}, {}, {}, {}};
  int k0 = kseg << 10;
  for (int kk = k0; kk < k0 + 1024; kk += 32) {
    __builtin_prefetch(Wr + kk + 32, 0, 0);        // stream the 134MB weight
    Frag a;
#pragma unroll
    for (int i = 0; i < 8; ++i) {
      int e = i << 1;
      int k = kk + ((e & 8) << 1) + half8 + (e & 7);
      float2 w2 = *(const float2*)(Wr + k);
      a.p[i] = pk(w2.x, w2.y);
    }
#pragma unroll
    for (int nt = 0; nt < 4; ++nt) {
      float fb[16];
#pragma unroll
      for (int e = 0; e < 16; ++e)
        fb[e] = flat[(size_t)((nt << 4) + mrow) * 65536 + kk + ((e & 8) << 1) + half8 + (e & 7)];
      Frag b;
#pragma unroll
      for (int i = 0; i < 8; ++i) b.p[i] = pk(fb[2 * i], fb[2 * i + 1]);
      acc[nt] = wmma_f16(a.v, b.v, acc[nt]);
    }
  }
#pragma unroll
  for (int nt = 0; nt < 4; ++nt)
#pragma unroll
    for (int v = 0; v < 8; ++v) {
      int m = (mt << 4) + v + half8;
      int n = (nt << 4) + mrow;
      atomicAdd(&hidden[(size_t)n * 512 + m], acc[nt][v]);
    }
}

// ---------------------------------------------------------------------------
// Regression layer 2: freq = sigmoid((hidden+b1) @ W2^T + b2), 64x6. VALU.
// ---------------------------------------------------------------------------
__global__ void reg2_k(const float* __restrict__ hidden,
                       const float* __restrict__ b1,
                       const float* __restrict__ W2,
                       const float* __restrict__ b2,
                       float* __restrict__ freq) {
  int t = threadIdx.x;
  if (t >= 384) return;
  int b = t / 6, j = t % 6;
  float s = b2[j];
  for (int h = 0; h < 512; ++h)
    s += (hidden[(size_t)b * 512 + h] + b1[h]) * W2[(size_t)j * 512 + h];
  freq[t] = sigmoidf(s);
}

// ---------------------------------------------------------------------------
// Final transposed conv (Cin=128, Cout=1, K=16, s2 p7) + lrelu + sigmoid. VALU.
// ---------------------------------------------------------------------------
__global__ void dect3_sig_k(const float* __restrict__ X,   // (64,128,2048)
                            const float* __restrict__ Wt,  // (128,1,16)
                            float* __restrict__ dec) {     // (64,1,4096)
  int tid = blockIdx.x * 256 + threadIdx.x;  // 262144
  int b = tid >> 12, l = tid & 4095;
  const float* Xb = X + (size_t)b * (128 * 2048);
  float s = 0.f;
  for (int ci = 0; ci < 128; ++ci) {
    const float* xr = Xb + ci * 2048;
    const float* wr = Wt + ci * 16;
#pragma unroll
    for (int t = 0; t < 16; ++t) {
      int pd = l - 8 + t;                    // pad_eff = 16-1-7 = 8
      if (pd >= 0 && pd < 4095 && (pd & 1) == 0)
        s += wr[15 - t] * xr[pd >> 1];
    }
  }
  dec[tid] = sigmoidf(lrelu(s));
}

// ---------------------------------------------------------------------------
extern "C" void kernel_launch(void* const* d_in, const int* in_sizes, int n_in,
                              void* d_out, int out_size, void* d_ws, size_t ws_size,
                              hipStream_t stream) {
  const float* x    = (const float*)d_in[0];
  const float* ew1  = (const float*)d_in[1];
  const float* ew2  = (const float*)d_in[2];
  const float* ew3  = (const float*)d_in[3];
  const float* prew = (const float*)d_in[4];
  const float* preb = (const float*)d_in[5];
  const float* emb  = (const float*)d_in[6];
  const float* rw1  = (const float*)d_in[7];
  const float* rb1  = (const float*)d_in[8];
  const float* rw2  = (const float*)d_in[9];
  const float* rb2  = (const float*)d_in[10];
  const float* dw0  = (const float*)d_in[11];
  const float* db0  = (const float*)d_in[12];
  const float* dtw1 = (const float*)d_in[13];
  const float* dtw2 = (const float*)d_in[14];
  const float* dtw3 = (const float*)d_in[15];

  float* out   = (float*)d_out;
  float* enc   = out;                    // 64*128*512 = 4194304
  float* perp  = out + 4194304;
  float* lossO = out + 4194305;
  float* freq  = out + 4194306;          // 64*6 = 384
  float* dec   = out + 4194690;          // 64*4096

  float* ws = (float*)d_ws;
  size_t o = 0;
  float* h1      = ws + o; o += 16777216;   // (64,128,2048)
  float* h2      = ws + o; o += 16777216;   // (64,256,1024)
  float* h3      = ws + o; o += 8388608;    // (64,256,512)
  float* z       = ws + o; o += 4194304;    // (64,128,512)
  float* hist    = ws + o; o += 1024;
  float* lossAcc = ws + o; o += 32;
  float* hidden  = ws + o; o += 32768;      // (64,512)
  float* enorm   = ws + o; o += 1024;
  int*   idx     = (int*)(ws + o); o += 32768;
  float* d0 = h3;   // reuse: h3 dead after pre-conv
  float* d1 = h1;   // reuse: h1 dead after conv2
  float* d2 = h2;   // reuse: h2 dead after conv3

  // zero accumulators (hist | lossAcc | hidden are contiguous)
  (void)hipMemsetAsync(hist, 0, (1024 + 32 + 32768) * sizeof(float), stream);

  emb_norm_k<<<4, 256, 0, stream>>>(emb, enorm);
  conv1_lrelu_k<<<65536, 256, 0, stream>>>(x, ew1, h1);

  // encoder GEMM convs. waves = (Cout/64)*(B*Lout/16), 8 waves/block, MT=4
  conv_gemm_k<8, 2, 3, 1, 4><<<2048, 256, 0, stream>>>(h1, ew2, nullptr, h2,
                                                       128, 256, 2048, 1024, 10, 65536);
  conv_gemm_k<4, 2, 1, 1, 4><<<1024, 256, 0, stream>>>(h2, ew3, nullptr, h3,
                                                       256, 256, 1024, 512, 9, 32768);
  conv_gemm_k<1, 1, 0, 0, 4><<<512, 256, 0, stream>>>(h3, prew, preb, z,
                                                      256, 128, 512, 512, 9, 32768);

  // vector quantizer
  vq_assign_k<<<256, 256, 0, stream>>>(z, emb, enorm, idx, hist);
  vq_gather_loss_k<<<16384, 256, 0, stream>>>(z, emb, idx, enc, lossAcc);
  vq_finalize_k<<<1, 1024, 0, stream>>>(hist, lossAcc, perp, lossO);

  // regression head
  reg1_gemm_k<<<256, 256, 0, stream>>>(rw1, enc, hidden);
  reg2_k<<<1, 384, 0, stream>>>(hidden, rb1, rw2, rb2, freq);

  // decoder
  conv_gemm_k<3, 1, 1, 0, 4><<<1024, 256, 0, stream>>>(enc, dw0, db0, d0,
                                                       128, 256, 512, 512, 9, 32768);
  convt_gemm_k<4, 2, 1, 1, 4><<<2048, 256, 0, stream>>>(d0, dtw1, d1,
                                                        256, 256, 512, 1024, 10, 65536);
  convt_gemm_k<8, 2, 3, 1, 4><<<2048, 256, 0, stream>>>(d1, dtw2, d2,
                                                        256, 128, 1024, 2048, 11, 131072);
  dect3_sig_k<<<1024, 256, 0, stream>>>(d2, dtw3, dec);

  (void)in_sizes; (void)n_in; (void)out_size; (void)ws_size;
}
